// HookedMambaBlock_84490596647003
// MI455X (gfx1250) — matrix operand order repaired and checked by reading, not compiled, not run on hardware
//
#include <hip/hip_runtime.h>
#include <stdint.h>

typedef __bf16 bf16_t;
typedef __attribute__((ext_vector_type(16))) bf16_t v16bf;
typedef __attribute__((ext_vector_type(8)))  float  v8f;

#define B_  2
#define L_  2048
#define D_  1024
#define E_  2048
#define N_  16
#define DR_ 64
#define DC_ 4
#define ML  (B_ * L_)   // 4096 rows

// ---------- helpers ----------
__device__ __forceinline__ unsigned short f2bf(float f) {
    unsigned u = __float_as_uint(f);
    unsigned r = (u + 0x7FFFu + ((u >> 16) & 1u)) >> 16;   // RNE
    return (unsigned short)r;
}
__device__ __forceinline__ float dsilu(float x) { return x / (1.0f + __expf(-x)); }

// ---------- tiny conversion kernels ----------
__global__ void k_f32_to_bf16(const float* __restrict__ src, unsigned short* __restrict__ dst, int n) {
    int i = blockIdx.x * blockDim.x + threadIdx.x;
    if (i < n) dst[i] = f2bf(src[i]);
}
__global__ void k_aneg(const float* __restrict__ a_log, float* __restrict__ a_neg, int n) {
    int i = blockIdx.x * blockDim.x + threadIdx.x;
    if (i < n) a_neg[i] = -__expf(a_log[i]);
}

// ---------- RMSNorm -> bf16 ----------
__global__ void k_rmsnorm(const float* __restrict__ resid, const float* __restrict__ w,
                          unsigned short* __restrict__ xn_bf) {
    __shared__ float red[8];
    int row = blockIdx.x;
    int t = threadIdx.x;                       // 256 threads, 4 floats each (D=1024)
    const float* r = resid + (size_t)row * D_;
    float4 v = ((const float4*)r)[t];
    float ss = v.x * v.x + v.y * v.y + v.z * v.z + v.w * v.w;
    #pragma unroll
    for (int m = 16; m >= 1; m >>= 1) ss += __shfl_xor(ss, m, 32);
    if ((t & 31) == 0) red[t >> 5] = ss;
    __syncthreads();
    float tot = 0.f;
    #pragma unroll
    for (int i = 0; i < 8; ++i) tot += red[i];
    float sc = rsqrtf(tot / (float)D_ + 1e-5f);
    float4 wv = ((const float4*)w)[t];
    unsigned short o0 = f2bf(v.x * sc * wv.x), o1 = f2bf(v.y * sc * wv.y);
    unsigned short o2 = f2bf(v.z * sc * wv.z), o3 = f2bf(v.w * sc * wv.w);
    uint2 pk; pk.x = (unsigned)o0 | ((unsigned)o1 << 16); pk.y = (unsigned)o2 | ((unsigned)o3 << 16);
    ((uint2*)(xn_bf + (size_t)row * D_))[t] = pk;
}

// ---------- bf16 WMMA GEMM: C[M,N] = A[M,K] * W[N,K]^T (+ addend) ----------
// Per-lane fragment layout used here (wave32; contiguous k, shared by A and B, not the ISA table order):
//   A 16x32 bf16 : lane l -> row m=l&15, half=l>>4; k runs {half*16 .. +15}
//   B 32x16 bf16 : lane l -> col n=l&15, half=l>>4; k runs {half*16 .. +15}
// Both K-contiguous in memory -> two b128 loads per fragment, no transpose needed.
// Wave tile: 32(M) x 64(N) = 8 WMMAs per K-step from 12 b128 loads (1.5 vmem/WMMA).
// Software double-buffered: next K-step's fragments are loaded before the current
// WMMAs consume theirs, so waits stay partial (no loadcnt==0 drains in the loop).
union FragBF { v16bf v; uint4 q[2]; };

#define LOAD_FRAGS(A0, A1, Bb0, Bb1, Bb2, Bb3, KK)                                  \
    do {                                                                            \
        A0.q[0]  = *(const uint4*)(aRow0 + (KK));                                   \
        A0.q[1]  = *(const uint4*)(aRow0 + (KK) + 8);                               \
        A1.q[0]  = *(const uint4*)(aRow1 + (KK));                                   \
        A1.q[1]  = *(const uint4*)(aRow1 + (KK) + 8);                               \
        Bb0.q[0] = *(const uint4*)(bRow0 + (KK));                                   \
        Bb0.q[1] = *(const uint4*)(bRow0 + (KK) + 8);                               \
        Bb1.q[0] = *(const uint4*)(bRow1 + (KK));                                   \
        Bb1.q[1] = *(const uint4*)(bRow1 + (KK) + 8);                               \
        Bb2.q[0] = *(const uint4*)(bRow2 + (KK));                                   \
        Bb2.q[1] = *(const uint4*)(bRow2 + (KK) + 8);                               \
        Bb3.q[0] = *(const uint4*)(bRow3 + (KK));                                   \
        Bb3.q[1] = *(const uint4*)(bRow3 + (KK) + 8);                               \
    } while (0)

#define DO_WMMAS(A0, A1, Bb0, Bb1, Bb2, Bb3)                                                              \
    do {                                                                                                  \
        acc00 = __builtin_amdgcn_wmma_f32_16x16x32_bf16(false, A0.v, false, Bb0.v, (short)0, acc00, false, false); \
        acc10 = __builtin_amdgcn_wmma_f32_16x16x32_bf16(false, A1.v, false, Bb0.v, (short)0, acc10, false, false); \
        acc01 = __builtin_amdgcn_wmma_f32_16x16x32_bf16(false, A0.v, false, Bb1.v, (short)0, acc01, false, false); \
        acc11 = __builtin_amdgcn_wmma_f32_16x16x32_bf16(false, A1.v, false, Bb1.v, (short)0, acc11, false, false); \
        acc02 = __builtin_amdgcn_wmma_f32_16x16x32_bf16(false, A0.v, false, Bb2.v, (short)0, acc02, false, false); \
        acc12 = __builtin_amdgcn_wmma_f32_16x16x32_bf16(false, A1.v, false, Bb2.v, (short)0, acc12, false, false); \
        acc03 = __builtin_amdgcn_wmma_f32_16x16x32_bf16(false, A0.v, false, Bb3.v, (short)0, acc03, false, false); \
        acc13 = __builtin_amdgcn_wmma_f32_16x16x32_bf16(false, A1.v, false, Bb3.v, (short)0, acc13, false, false); \
    } while (0)

__global__ void __launch_bounds__(256) k_gemm_bf16(const unsigned short* __restrict__ A,
                            const unsigned short* __restrict__ W,
                            float* __restrict__ C,
                            const float* __restrict__ addend,
                            int M, int N, int K) {
    int lane = threadIdx.x & 31;
    int wave = threadIdx.x >> 5;               // 8 waves
    int m32   = blockIdx.y * 256 + wave * 32;  // wave's 32-row M tile
    int nBase = blockIdx.x * 64;               // block's N strip (4x16)
    int hf  = lane >> 4;
    int l15 = lane & 15;

    const unsigned short* aRow0 = A + (size_t)(m32      + l15) * K + hf * 16;
    const unsigned short* aRow1 = A + (size_t)(m32 + 16 + l15) * K + hf * 16;
    const unsigned short* bRow0 = W + (size_t)(nBase +  0 + l15) * K + hf * 16;
    const unsigned short* bRow1 = W + (size_t)(nBase + 16 + l15) * K + hf * 16;
    const unsigned short* bRow2 = W + (size_t)(nBase + 32 + l15) * K + hf * 16;
    const unsigned short* bRow3 = W + (size_t)(nBase + 48 + l15) * K + hf * 16;

    v8f acc00 = {}, acc01 = {}, acc02 = {}, acc03 = {};
    v8f acc10 = {}, acc11 = {}, acc12 = {}, acc13 = {};

    FragBF a0, a1, b0, b1, b2, b3;
    LOAD_FRAGS(a0, a1, b0, b1, b2, b3, 0);

    for (int kk = 32; kk < K; kk += 32) {
        FragBF na0, na1, nb0, nb1, nb2, nb3;
        LOAD_FRAGS(na0, na1, nb0, nb1, nb2, nb3, kk);   // prefetch next K-step
        DO_WMMAS(a0, a1, b0, b1, b2, b3);               // consume current
        a0 = na0; a1 = na1;
        b0 = nb0; b1 = nb1; b2 = nb2; b3 = nb3;
    }
    DO_WMMAS(a0, a1, b0, b1, b2, b3);                   // epilogue K-step

    // D layout: VGPR v, lane l -> row = v + (l>>4)*8, col = l&15
    v8f accs[2][4] = { { acc00, acc01, acc02, acc03 }, { acc10, acc11, acc12, acc13 } };
    #pragma unroll
    for (int g = 0; g < 2; ++g) {
        int rowBase = m32 + g * 16 + hf * 8;
        #pragma unroll
        for (int t = 0; t < 4; ++t) {
            int col = nBase + t * 16 + l15;
            #pragma unroll
            for (int v = 0; v < 8; ++v) {
                size_t idx = (size_t)(rowBase + v) * N + col;
                float val = accs[g][t][v];
                if (addend) val += addend[idx];
                C[idx] = val;
            }
        }
    }
}

// ---------- causal depthwise conv1d + SiLU ----------
__global__ void k_conv_silu(const float* __restrict__ xin, const float* __restrict__ cw,
                            const float* __restrict__ cb, float* __restrict__ xs) {
    int idx = blockIdx.x * blockDim.x + threadIdx.x;     // over ML*E
    int e = idx % E_;
    int l = (idx / E_) % L_;
    int b = idx / (E_ * L_);
    const float* base = xin + (size_t)b * L_ * E_ + e;
    float acc = cb[e];
    #pragma unroll
    for (int k = 0; k < DC_; ++k) {
        int ls = l + k - (DC_ - 1);
        if (ls >= 0) acc += base[(size_t)ls * E_] * cw[e * DC_ + k];
    }
    xs[idx] = dsilu(acc);
}

// ---------- per-position projections: delta (softplus), Bm, Cm ----------
__global__ void k_proj(const float* __restrict__ xs, const float* __restrict__ wd1,
                       const float* __restrict__ wd2, const float* __restrict__ wd2b,
                       const float* __restrict__ wB, const float* __restrict__ wC,
                       float* __restrict__ delta, float* __restrict__ Bm, float* __restrict__ Cm) {
    __shared__ float xrow[E_];
    __shared__ float dr[DR_];
    int pos = blockIdx.x;                     // 0..ML-1
    int t = threadIdx.x;                      // 256
    const float* xr = xs + (size_t)pos * E_;
    for (int i = t; i < E_; i += 256) xrow[i] = xr[i];
    __syncthreads();
    // dr[j] = x . wd1[j,:]   (4 lanes per j, shfl-reduce within the quad)
    {
        int j = t >> 2, q = t & 3;
        const float* wr = wd1 + (size_t)j * E_;
        float s = 0.f;
        for (int e = q * 512; e < q * 512 + 512; ++e) s += xrow[e] * wr[e];
        s += __shfl_xor(s, 1, 32);
        s += __shfl_xor(s, 2, 32);
        if (q == 0) dr[j] = s;
    }
    // Bm / Cm (16 + 16 dots)
    if (t < 128) {
        int j = t >> 2, q = t & 3;
        const float* wr = (j < 16) ? (wB + (size_t)j * E_) : (wC + (size_t)(j - 16) * E_);
        float s = 0.f;
        for (int e = q * 512; e < q * 512 + 512; ++e) s += xrow[e] * wr[e];
        s += __shfl_xor(s, 1, 32);
        s += __shfl_xor(s, 2, 32);
        if (q == 0) {
            if (j < 16) Bm[(size_t)pos * N_ + j] = s;
            else        Cm[(size_t)pos * N_ + (j - 16)] = s;
        }
    }
    __syncthreads();
    // delta[e] = softplus(dr . wd2[e,:] + wd2_b[e])
    for (int e = t; e < E_; e += 256) {
        const float* w2 = wd2 + (size_t)e * DR_;
        float s = wd2b[e];
        #pragma unroll
        for (int j = 0; j < DR_; ++j) s += dr[j] * w2[j];
        delta[(size_t)pos * E_ + e] = (s > 20.f) ? s : log1pf(__expf(s));
    }
}

// ---------- selective scan: one lane per (b,e,n); n-dot via 16-lane shfl tree ----------
__global__ void k_scan(const float* __restrict__ xs, const float* __restrict__ delta,
                       const float* __restrict__ Bm, const float* __restrict__ Cm,
                       const float* __restrict__ Aneg, float* __restrict__ y) {
    int t = threadIdx.x;                      // 256: 16 e-values x 16 n
    int n = t & 15;
    int eg = blockIdx.x % (E_ / 16);
    int b  = blockIdx.x / (E_ / 16);
    int e = eg * 16 + (t >> 4);
    float a = Aneg[(size_t)e * N_ + n];
    float h = 0.f;
    size_t rowE = (size_t)b * L_ * E_ + e;
    size_t rowN = (size_t)b * L_ * N_ + n;
    for (int l = 0; l < L_; ++l) {
        float xt = xs[rowE + (size_t)l * E_];
        float dt = delta[rowE + (size_t)l * E_];
        float bn = Bm[rowN + (size_t)l * N_];
        float cn = Cm[rowN + (size_t)l * N_];
        h = __expf(dt * a) * h + dt * xt * bn;
        float p = h * cn;
        p += __shfl_xor(p, 1, 16);
        p += __shfl_xor(p, 2, 16);
        p += __shfl_xor(p, 4, 16);
        p += __shfl_xor(p, 8, 16);
        if (n == 0) y[rowE + (size_t)l * E_] = p;
    }
}

// ---------- gate: y = (y + W_D*x) * silu(skip)  -> bf16 ----------
__global__ void k_gate(const float* __restrict__ y, const float* __restrict__ xs,
                       const float* __restrict__ skip, const float* __restrict__ WD,
                       unsigned short* __restrict__ ybf) {
    int idx = blockIdx.x * blockDim.x + threadIdx.x;
    int e = idx % E_;
    float v = (y[idx] + WD[e] * xs[idx]) * dsilu(skip[idx]);
    ybf[idx] = f2bf(v);
}

// ---------- launch ----------
extern "C" void kernel_launch(void* const* d_in, const int* in_sizes, int n_in,
                              void* d_out, int out_size, void* d_ws, size_t ws_size,
                              hipStream_t stream) {
    const float* resid  = (const float*)d_in[0];
    const float* norm_w = (const float*)d_in[1];
    const float* skip_w = (const float*)d_in[2];
    const float* in_w   = (const float*)d_in[3];
    const float* conv_w = (const float*)d_in[4];
    const float* conv_b = (const float*)d_in[5];
    const float* wd1    = (const float*)d_in[6];
    const float* wd2    = (const float*)d_in[7];
    const float* wd2b   = (const float*)d_in[8];
    const float* wB     = (const float*)d_in[9];
    const float* wC     = (const float*)d_in[10];
    const float* A_log  = (const float*)d_in[11];
    const float* W_D    = (const float*)d_in[12];
    const float* out_w  = (const float*)d_in[13];

    char* ws = (char*)d_ws;
    size_t off = 0;
    auto alloc = [&](size_t bytes) -> void* {
        void* p = ws + off;
        off = (off + bytes + 255) & ~(size_t)255;
        return p;
    };
    unsigned short* xn_bf   = (unsigned short*)alloc((size_t)ML * D_ * 2);
    unsigned short* skipw_bf= (unsigned short*)alloc((size_t)E_ * D_ * 2);
    unsigned short* inw_bf  = (unsigned short*)alloc((size_t)E_ * D_ * 2);
    unsigned short* outw_bf = (unsigned short*)alloc((size_t)D_ * E_ * 2);
    float* skip_f = (float*)alloc((size_t)ML * E_ * 4);
    float* xin_f  = (float*)alloc((size_t)ML * E_ * 4);   // reused as y_f after conv
    float* xsilu  = (float*)alloc((size_t)ML * E_ * 4);
    float* delta  = (float*)alloc((size_t)ML * E_ * 4);
    float* Bm     = (float*)alloc((size_t)ML * N_ * 4);
    float* Cm     = (float*)alloc((size_t)ML * N_ * 4);
    float* Aneg   = (float*)alloc((size_t)E_ * N_ * 4);
    unsigned short* y_bf = (unsigned short*)alloc((size_t)ML * E_ * 2);
    float* y_f = xin_f;

    k_f32_to_bf16<<<(E_ * D_ + 255) / 256, 256, 0, stream>>>(skip_w, skipw_bf, E_ * D_);
    k_f32_to_bf16<<<(E_ * D_ + 255) / 256, 256, 0, stream>>>(in_w,  inw_bf,  E_ * D_);
    k_f32_to_bf16<<<(D_ * E_ + 255) / 256, 256, 0, stream>>>(out_w, outw_bf, D_ * E_);
    k_aneg<<<(E_ * N_ + 255) / 256, 256, 0, stream>>>(A_log, Aneg, E_ * N_);

    k_rmsnorm<<<ML, 256, 0, stream>>>(resid, norm_w, xn_bf);

    dim3 g1(E_ / 64, ML / 256);   // (32,16)
    k_gemm_bf16<<<g1, 256, 0, stream>>>(xn_bf, skipw_bf, skip_f, nullptr, ML, E_, D_);
    k_gemm_bf16<<<g1, 256, 0, stream>>>(xn_bf, inw_bf,  xin_f,  nullptr, ML, E_, D_);

    k_conv_silu<<<(ML * E_) / 256, 256, 0, stream>>>(xin_f, conv_w, conv_b, xsilu);
    k_proj<<<ML, 256, 0, stream>>>(xsilu, wd1, wd2, wd2b, wB, wC, delta, Bm, Cm);
    k_scan<<<B_ * (E_ / 16), 256, 0, stream>>>(xsilu, delta, Bm, Cm, Aneg, y_f);
    k_gate<<<(ML * E_) / 256, 256, 0, stream>>>(y_f, xsilu, skip_f, W_D, y_bf);

    dim3 g2(D_ / 64, ML / 256);   // (16,16)
    k_gemm_bf16<<<g2, 256, 0, stream>>>(y_bf, outw_bf, (float*)d_out, resid, ML, D_, E_);
}
